// NoCeptionNet_15582141350043
// MI455X (gfx1250) — matrix operand order repaired and checked by reading, not compiled, not run on hardware
//
#include <hip/hip_runtime.h>
#include <hip/hip_bf16.h>

// ---------------------------------------------------------------------------
// NoCeptionNet on MI455X (gfx1250): fused NNConv-max GNN.
// Transposed WMMA formulation: D = W_chunk^T (16x32) x efeat^T (32x16) via
// v_wmma_f32_16x16x32_bf16, so D columns are edges -> per-lane msg scaling
// and scatter-max with no cross-lane shuffles and no exec divergence.
// ---------------------------------------------------------------------------

typedef __attribute__((ext_vector_type(16))) __bf16 v16bf;
typedef __attribute__((ext_vector_type(8)))  float  v8f;

#define HD 32     // hidden dim
#define CW 512    // H * H/2
#define NLAYER 2

__device__ __forceinline__ float elu1(float x) { return x > 0.f ? x : expm1f(x); }

// branchless monotonic unsigned key for float ordering (u32 atomicMax == fmax)
__device__ __forceinline__ unsigned f2k(float f) {
    unsigned u = __float_as_uint(f);
    return u ^ ((unsigned)((int)u >> 31) | 0x80000000u);
}
__device__ __forceinline__ float k2f(unsigned k) {
    return __uint_as_float(k ^ ((unsigned)((int)(~k) >> 31) | 0x80000000u));
}
// float -> bf16 bits, round to nearest even
__device__ __forceinline__ unsigned short f2bf(float f) {
    unsigned u = __float_as_uint(f);
    u += 0x7fffu + ((u >> 16) & 1u);
    return (unsigned short)(u >> 16);
}

// ---------------- encoders: h = elu(x @ W + b) -----------------------------
__global__ void k_encode(const float* __restrict__ node_inp,
                         const float* __restrict__ edge_inp,
                         const float* __restrict__ Wn, const float* __restrict__ bn,
                         const float* __restrict__ We, const float* __restrict__ be,
                         float* __restrict__ h_n, float* __restrict__ h_e,
                         int N, int E) {
    int t = blockIdx.x * blockDim.x + threadIdx.x;
    int f = t & 31;
    int r = t >> 5;
    if (r < N) {
        float v = node_inp[r * 2] * Wn[f] + node_inp[r * 2 + 1] * Wn[HD + f] + bn[f];
        h_n[r * HD + f] = elu1(v);
    }
    if (r < E) {
        float v = edge_inp[r] * We[f] + be[f];
        h_e[r * HD + f] = elu1(v);
    }
}

// ------ pack miW/moW into per-lane WMMA *A* layout (transposed chunk) ------
// A(16x32) for chunk cc:  A[j][k] = W[k][cc*16 + j]   (j = output feature)
// A layout used here (contiguous k, same as B): lane (khalf,nn) element e -> row j = nn,
//                  K = khalf*16 + e
// packed[((l*32 + cc)*32 + lane)*16 + e] = W[l][K*512 + cc*16 + nn]
__global__ void k_pack_w(const float* __restrict__ miW, const float* __restrict__ moW,
                         unsigned short* __restrict__ miP, unsigned short* __restrict__ moP) {
    int t = blockIdx.x * blockDim.x + threadIdx.x;
    if (t >= NLAYER * 32 * 32 * 16) return;
    int e    = t & 15;
    int lane = (t >> 4) & 31;
    int cc   = (t >> 9) & 31;
    int l    = t >> 14;
    int khalf = lane >> 4, nn = lane & 15;
    int k = khalf * 16 + e;
    size_t si = (size_t)l * (HD * CW) + (size_t)k * CW + cc * 16 + nn;
    miP[t] = f2bf(miW[si]);
    moP[t] = f2bf(moW[si]);
}

// ---------------- layernorm (one wave32 per row of 32) ---------------------
__global__ void k_ln_nodes(const float* __restrict__ h, const float* __restrict__ g,
                           const float* __restrict__ b, float* __restrict__ out,
                           int rows, int layer) {
    int wave = (blockIdx.x * blockDim.x + threadIdx.x) >> 5;
    int lane = threadIdx.x & 31;
    if (wave >= rows) return;
    float x = h[(size_t)wave * HD + lane];
    float s = x;
    for (int o = 16; o; o >>= 1) s += __shfl_xor(s, o, 32);
    float m = s * (1.f / 32.f);
    float d = x - m;
    float vv = d * d;
    for (int o = 16; o; o >>= 1) vv += __shfl_xor(vv, o, 32);
    float y = d * rsqrtf(vv * (1.f / 32.f) + 1e-5f) * g[layer * HD + lane] + b[layer * HD + lane];
    out[(size_t)wave * HD + lane] = y;
}

__global__ void k_ln_edges(const float* __restrict__ h, const float* __restrict__ g,
                           const float* __restrict__ b, unsigned short* __restrict__ out,
                           int rows, int layer) {
    int wave = (blockIdx.x * blockDim.x + threadIdx.x) >> 5;
    int lane = threadIdx.x & 31;
    if (wave >= rows) return;
    float x = h[(size_t)wave * HD + lane];
    float s = x;
    for (int o = 16; o; o >>= 1) s += __shfl_xor(s, o, 32);
    float m = s * (1.f / 32.f);
    float d = x - m;
    float vv = d * d;
    for (int o = 16; o; o >>= 1) vv += __shfl_xor(vv, o, 32);
    float y = d * rsqrtf(vv * (1.f / 32.f) + 1e-5f) * g[layer * HD + lane] + b[layer * HD + lane];
    out[(size_t)wave * HD + lane] = f2bf(y);
}

// ---------------- fill u32 buffer ------------------------------------------
__global__ void k_fill(unsigned* __restrict__ p, size_t n, unsigned v) {
    size_t t = (size_t)blockIdx.x * blockDim.x + threadIdx.x;
    size_t stride = (size_t)gridDim.x * blockDim.x;
    for (; t < n; t += stride) p[t] = v;
}

// ---------------- fused edge conv: WMMA GEMM + scatter max -----------------
// One wave32 per 16-edge tile. For each 16-row chunk cc (== input-feature i):
//   D(16x16) = Wchunk^T(16x32 bf16) x efeat_tile^T(32x16 bf16) + bias
//   D[j][m] = Wi[edge m][cc*16+j];  msg = nfeat[srcnode(m)][cc] * D[j][m]
//   -> atomicMax(key) into agg[dstnode(m)*512 + cc*16 + j]
// Lane owns edge m = lane&15 (both half-waves), reg r owns j = r + 8*(lane>>4).
__global__ void __launch_bounds__(256)
k_edge_conv(const unsigned short* __restrict__ efeat,   // E x 32 bf16 (rows)
            const float* __restrict__ nfeat,            // N x 32 f32
            const int* __restrict__ src, const int* __restrict__ dst,
            const unsigned short* __restrict__ wiP,     // layer-offset packed A
            const unsigned short* __restrict__ woP,
            const float* __restrict__ mib, const float* __restrict__ mob, // layer-offset (512)
            unsigned* __restrict__ aggI, unsigned* __restrict__ aggO,
            int N, int E) {
    int wave = threadIdx.x >> 5;
    int tile = blockIdx.x * (blockDim.x >> 5) + wave;
    int tb = tile * 16;
    if (tb >= E) return;                       // wave-uniform guard (WMMA needs full EXEC)

    int lane  = threadIdx.x & 31;
    int khalf = lane >> 4;
    int nn    = lane & 15;
    int lim   = E - tb;
    bool valid = (nn < lim);
    int ei = tb + (valid ? nn : lim - 1);      // clamped edge index for loads

    // ---- B matrix: efeat^T, lane = edge nn, K = khalf*16 + e (contiguous) ----
    const unsigned* erow = (const unsigned*)(efeat + (size_t)ei * HD);
    union { v16bf v; unsigned u[8]; } B;
#pragma unroll
    for (int p = 0; p < 8; p++) B.u[p] = erow[khalf * 8 + p];

    int sN = src[ei], dN = dst[ei];
    const float* hIrow = nfeat + (size_t)sN * HD;  // mi: messages scaled by nfeat[src]
    const float* hOrow = nfeat + (size_t)dN * HD;  // mo: messages scaled by nfeat[dst]
    // invalid lanes scatter into dump row N (never read back) -> no divergence
    unsigned* aI = aggI + (size_t)(valid ? dN : N) * CW + 8 * khalf;  // mi aggregates at dst
    unsigned* aO = aggO + (size_t)(valid ? sN : N) * CW + 8 * khalf;  // mo aggregates at src

    for (int cc = 0; cc < 32; ++cc) {
        union { v16bf v; unsigned u[8]; } Ai, Ao;
        const unsigned* ai = (const unsigned*)(wiP + ((size_t)(cc * 32 + lane)) * 16);
        const unsigned* ao = (const unsigned*)(woP + ((size_t)(cc * 32 + lane)) * 16);
#pragma unroll
        for (int p = 0; p < 8; p++) { Ai.u[p] = ai[p]; Ao.u[p] = ao[p]; }

        v8f Ci, Co;
#pragma unroll
        for (int r = 0; r < 8; r++) {
            int j = cc * 16 + 8 * khalf + r;    // broadcast within half-wave (L0 hit)
            Ci[r] = mib[j];
            Co[r] = mob[j];
        }

        v8f Di = __builtin_amdgcn_wmma_f32_16x16x32_bf16(false, Ai.v, false, B.v,
                                                         (short)0, Ci, false, false);
        v8f Do = __builtin_amdgcn_wmma_f32_16x16x32_bf16(false, Ao.v, false, B.v,
                                                         (short)0, Co, false, false);

        float hI = hIrow[cc];
        float hO = hOrow[cc];
        unsigned* pI = aI + cc * 16;
        unsigned* pO = aO + cc * 16;
#pragma unroll
        for (int r = 0; r < 8; r++) {
            atomicMax(pI + r, f2k(hI * Di[r]));
            atomicMax(pO + r, f2k(hO * Do[r]));
        }
    }
}

// ---------------- node update: h = elu(nfeat + [mi, mo]) -------------------
__global__ void k_node_update(const float* __restrict__ nfeat,
                              const unsigned* __restrict__ aggI,
                              const unsigned* __restrict__ aggO,
                              const float* __restrict__ micb, const float* __restrict__ mocb,
                              float* __restrict__ h_n, int N) {
    int wave = (blockIdx.x * blockDim.x + threadIdx.x) >> 5;
    int lane = threadIdx.x & 31;
    if (wave >= N) return;
    int j = lane & 15;
    const unsigned* base = ((lane < 16) ? aggI : aggO) + (size_t)wave * CW + j;
    float acc = (lane < 16) ? micb[j] : mocb[j];
#pragma unroll
    for (int i = 0; i < 32; i++) {
        unsigned k = base[i * 16];
        acc += (k != 0u) ? k2f(k) : 0.f;   // key 0 == untouched (empty segment -> 0)
    }
    h_n[(size_t)wave * HD + lane] = elu1(nfeat[(size_t)wave * HD + lane] + acc);
}

// ---------------- readout max over nodes -----------------------------------
__global__ void k_rmax(const float* __restrict__ h_n, unsigned* __restrict__ slots, int N) {
    int gt = blockIdx.x * blockDim.x + threadIdx.x;
    int f = gt & 31;
    int row0 = gt >> 5;
    int rows = (gridDim.x * blockDim.x) >> 5;
    bool any = false;
    float m = 0.f;
    for (int n = row0; n < N; n += rows) {
        float v = h_n[(size_t)n * HD + f];
        m = any ? fmaxf(m, v) : v;
        any = true;
    }
    if (any) atomicMax(slots + f, f2k(m));
}

// ---------------- MLP head: (1x33) @ (33x128) elu @ (128x1) ----------------
__global__ void k_mlp(const unsigned* __restrict__ slots, const float* __restrict__ gfeat,
                      const float* __restrict__ W1, const float* __restrict__ b1,
                      const float* __restrict__ W2, const float* __restrict__ b2,
                      float* __restrict__ out) {
    __shared__ float gv[33];
    __shared__ float partial[4];
    int t = threadIdx.x;  // 128 threads
    if (t < 32) gv[t] = k2f(slots[t]);
    if (t == 0) gv[32] = gfeat[0];
    __syncthreads();
    float acc = b1[t];
#pragma unroll
    for (int k = 0; k < 33; k++) acc += gv[k] * W1[k * 128 + t];
    float p = elu1(acc) * W2[t];
    for (int o = 16; o; o >>= 1) p += __shfl_xor(p, o, 32);
    if ((t & 31) == 0) partial[t >> 5] = p;
    __syncthreads();
    if (t == 0) out[0] = partial[0] + partial[1] + partial[2] + partial[3] + b2[0];
}

// ---------------------------------------------------------------------------
extern "C" void kernel_launch(void* const* d_in, const int* in_sizes, int n_in,
                              void* d_out, int out_size, void* d_ws, size_t ws_size,
                              hipStream_t stream) {
    const float* node_inp = (const float*)d_in[0];
    const float* edge_inp = (const float*)d_in[1];
    const float* graph_feat = (const float*)d_in[2];
    const int* src = (const int*)d_in[3];
    const int* dst = (const int*)d_in[4];
    const float* W_node = (const float*)d_in[5];
    const float* b_node = (const float*)d_in[6];
    const float* W_edge = (const float*)d_in[7];
    const float* b_edge = (const float*)d_in[8];
    const float* ln_g = (const float*)d_in[9];
    const float* ln_b = (const float*)d_in[10];
    const float* miW = (const float*)d_in[11];
    const float* mib = (const float*)d_in[12];
    const float* moW = (const float*)d_in[13];
    const float* mob = (const float*)d_in[14];
    const float* mi_cb = (const float*)d_in[15];
    const float* mo_cb = (const float*)d_in[16];
    const float* W1 = (const float*)d_in[17];
    const float* b1 = (const float*)d_in[18];
    const float* W2 = (const float*)d_in[19];
    const float* b2 = (const float*)d_in[20];

    const int N = in_sizes[0] / 2;
    const int E = in_sizes[1];

    // ---- workspace carve-out (256B aligned) ----
    char* ws = (char*)d_ws;
    size_t off = 0;
    auto carve = [&](size_t bytes) -> void* {
        void* p = ws + off;
        off = (off + bytes + 255) & ~(size_t)255;
        return p;
    };
    float* h_n            = (float*)carve((size_t)N * HD * 4);
    float* h_e            = (float*)carve((size_t)E * HD * 4);
    float* nfeat          = (float*)carve((size_t)N * HD * 4);
    unsigned short* efeat = (unsigned short*)carve((size_t)E * HD * 2);
    unsigned short* miP   = (unsigned short*)carve((size_t)NLAYER * 32 * 32 * 16 * 2);
    unsigned short* moP   = (unsigned short*)carve((size_t)NLAYER * 32 * 32 * 16 * 2);
    unsigned* aggI        = (unsigned*)carve((size_t)(N + 1) * CW * 4);  // +1 dump row
    unsigned* aggO        = (unsigned*)carve((size_t)(N + 1) * CW * 4);
    unsigned* slots       = (unsigned*)carve(32 * 4);

    const int RE = (N > E) ? N : E;
    k_encode<<<(RE * 32 + 255) / 256, 256, 0, stream>>>(node_inp, edge_inp, W_node, b_node,
                                                        W_edge, b_edge, h_n, h_e, N, E);
    k_pack_w<<<(NLAYER * 32 * 32 * 16 + 255) / 256, 256, 0, stream>>>(miW, moW, miP, moP);

    const size_t aggn = (size_t)N * CW;
    const int tiles = (E + 15) / 16;
    for (int l = 0; l < NLAYER; l++) {
        k_ln_nodes<<<(N * 32 + 255) / 256, 256, 0, stream>>>(h_n, ln_g, ln_b, nfeat, N, l);
        k_ln_edges<<<(E * 32 + 255) / 256, 256, 0, stream>>>(h_e, ln_g, ln_b, efeat, E, l);
        k_fill<<<2048, 256, 0, stream>>>(aggI, aggn, 0u);
        k_fill<<<2048, 256, 0, stream>>>(aggO, aggn, 0u);
        k_edge_conv<<<(tiles + 7) / 8, 256, 0, stream>>>(
            efeat, nfeat, src, dst,
            miP + (size_t)l * 32 * 32 * 16, moP + (size_t)l * 32 * 32 * 16,
            mib + (size_t)l * CW, mob + (size_t)l * CW, aggI, aggO, N, E);
        k_node_update<<<(N * 32 + 255) / 256, 256, 0, stream>>>(
            nfeat, aggI, aggO, mi_cb + l * 16, mo_cb + l * 16, h_n, N);
    }
    k_fill<<<1, 32, 0, stream>>>(slots, 32, 0u);
    k_rmax<<<64, 256, 0, stream>>>(h_n, slots, N);
    k_mlp<<<1, 128, 0, stream>>>(slots, graph_feat, W1, b1, W2, b2, (float*)d_out);
}
